// StemBlock_41377714930099
// MI455X (gfx1250) — matrix operand order repaired and checked by reading, not compiled, not run on hardware
//
#include <hip/hip_runtime.h>

typedef unsigned short ushort;
typedef unsigned int   uint;
typedef __attribute__((ext_vector_type(16))) __bf16 v16bf;
typedef __attribute__((ext_vector_type(8)))  float  v8f;
typedef __attribute__((ext_vector_type(8)))  ushort ushort8;
typedef __attribute__((ext_vector_type(4)))  uint   uint4v;

#define B_      16
#define CIN_    16
#define HW_     512
#define COUT_   64
#define OHW_    256
#define KPADMAX 800   // ceil(16*7*7 / 32)*32 ; fixed packed-weight stride
#define NTILE   128
#define KROW    40    // 32 + 8 pad: 80-byte LDS row stride -> conflict-light

union BFrag { ushort8 h[2]; v16bf v; };
union KT4   { int4 v[4]; int s[16]; };

__device__ __forceinline__ ushort f2bf(float f) {
  unsigned u = __float_as_uint(f);
  unsigned r = u + 0x7FFFu + ((u >> 16) & 1u);
  return (ushort)(r >> 16);
}
__device__ __forceinline__ uint pack2bf(float lo, float hi) {
  unsigned ul = __float_as_uint(lo); ul += 0x7FFFu + ((ul >> 16) & 1u);
  unsigned uh = __float_as_uint(hi); uh += 0x7FFFu + ((uh >> 16) & 1u);
  return (ul >> 16) | (uh & 0xFFFF0000u);
}

// ---------------- 1) global max-abs over the sliced weight window ----------
__global__ __launch_bounds__(256) void stem_scale_kernel(
    const float* __restrict__ w, const int* __restrict__ sub_path,
    float* __restrict__ scale_out) {
  __shared__ float red[256];
  const int ks = 3 + 2 * sub_path[1];      // KS_LIST = [3,5,7]
  const int s  = (7 - ks) >> 1;
  const int total = COUT_ * CIN_ * ks * ks;
  float m = 0.f;
  for (int idx = threadIdx.x; idx < total; idx += 256) {
    int kw = idx % ks; int t2 = idx / ks;
    int kh = t2 % ks;  int t3 = t2 / ks;
    int ci = t3 % CIN_; int co = t3 / CIN_;
    float v = w[((co * CIN_ + ci) * 7 + (s + kh)) * 7 + (s + kw)];
    m = fmaxf(m, fabsf(v));
  }
  red[threadIdx.x] = m;
  __syncthreads();
  for (int off = 128; off > 0; off >>= 1) {
    if (threadIdx.x < off)
      red[threadIdx.x] = fmaxf(red[threadIdx.x], red[threadIdx.x + off]);
    __syncthreads();
  }
  if (threadIdx.x == 0) scale_out[0] = fmaxf(red[0] / 127.f, 1e-30f);
}

// ---------------- 2) fake-quant + slice + pack weights to bf16 GEMM-A -----
// Unified dense layout over ALL 16 input channels (k = cin*ks*ks + kh*ks + kw);
// channels outside the cout-row's group are packed as zero so groups=1/2 both
// become one dense GEMM with one shared im2col patch per block.
__global__ __launch_bounds__(256) void stem_pack_kernel(
    const float* __restrict__ w, const int* __restrict__ sub_path,
    const float* __restrict__ scale_p, ushort* __restrict__ wbf) {
  int idx = blockIdx.x * 256 + threadIdx.x;
  if (idx >= COUT_ * KPADMAX) return;
  const int co = idx / KPADMAX;
  const int kp = idx - co * KPADMAX;
  const int ks     = 3 + 2 * sub_path[1];
  const int groups = 1 + sub_path[2];      // GROUPS_LIST = [1,2]
  const int cing = CIN_ / groups;
  const int ksq  = ks * ks;
  const int K    = CIN_ * ksq;
  float val = 0.f;
  if (kp < K) {
    int ci  = kp / ksq;
    int rem = kp - ci * ksq;
    int kh = rem / ks;
    int kw = rem - kh * ks;
    int g  = co / (COUT_ / groups);
    if (ci >= g * cing && ci < (g + 1) * cing) {
      int s  = (7 - ks) >> 1;
      float wv = w[((co * CIN_ + ci) * 7 + (s + kh)) * 7 + (s + kw)];
      float sc = scale_p[0];
      float q  = rintf(wv / sc);
      q = fminf(fmaxf(q, -127.f), 127.f);
      val = q * sc;
    }
  }
  wbf[idx] = f2bf(val);
}

// ---------------- 3) implicit-GEMM conv + fused BN/bias/PReLU/bias --------
// Block = 8 wave32; tile = 64 cout x 128 positions. Double-buffered LDS patch,
// one barrier per K-chunk. Pipeline order inside each iteration:
//   A/B frag loads -> staging loads (next chunk) -> 4 WMMAs -> pack+ds_store.
// In-order load completion lets WMMAs proceed on a partial loadcnt wait while
// the 16 staging loads are still in flight.
struct StageCtx {
  const float* xb;   // x + bb*CIN*HW*HW
  const int*   ktab; // LUT: base | kw<<23 | ok<<26  (row math pre-folded)
  int ns, kbase, iwb;
};

__device__ __forceinline__ void stage_load(const StageCtx& c, int kc,
                                           float vals[16], uint& okbits) {
  KT4 kt;
  const int4* ktp = (const int4*)(c.ktab + kc + c.kbase);
  kt.v[0] = ktp[0]; kt.v[1] = ktp[1]; kt.v[2] = ktp[2]; kt.v[3] = ktp[3];
  okbits = 0;
#pragma unroll
  for (int i = 0; i < 16; ++i) {
    int info = kt.s[i];
    int kw = (info >> 23) & 7;
    int iw = c.iwb + kw;
    uint ok = ((uint)(info >> 26) & 1u) & ((unsigned)iw < HW_ ? 1u : 0u);
    okbits |= ok << i;
    int iwc = iw < 0 ? 0 : (iw > HW_ - 1 ? HW_ - 1 : iw);
    vals[i] = c.xb[(info & 0x3FFFFF) + iwc];          // always-safe address
  }
}

__device__ __forceinline__ void stage_store(ushort* __restrict__ buf,
                                            const StageCtx& c,
                                            const float vals[16], uint okbits) {
  uint4v r0, r1;
#pragma unroll
  for (int i = 0; i < 4; ++i) {
    r0[i] = pack2bf((okbits >> (2 * i))     & 1u ? vals[2 * i]     : 0.f,
                    (okbits >> (2 * i + 1)) & 1u ? vals[2 * i + 1] : 0.f);
    r1[i] = pack2bf((okbits >> (2 * i + 8)) & 1u ? vals[2 * i + 8] : 0.f,
                    (okbits >> (2 * i + 9)) & 1u ? vals[2 * i + 9] : 0.f);
  }
  *(uint4v*)(buf + c.ns * KROW + c.kbase)     = r0;
  *(uint4v*)(buf + c.ns * KROW + c.kbase + 8) = r1;
}

__global__ __launch_bounds__(256) void stem_conv_kernel(
    const float* __restrict__ x, const ushort* __restrict__ wbf,
    const int* __restrict__ sub_path,
    const float* __restrict__ gamma, const float* __restrict__ beta,
    const float* __restrict__ mean,  const float* __restrict__ var,
    const float* __restrict__ b1,    const float* __restrict__ slope,
    const float* __restrict__ b2,    float* __restrict__ out) {
  __shared__ __attribute__((aligned(16))) ushort tileA[NTILE * KROW]; // 10 KB
  __shared__ __attribute__((aligned(16))) ushort tileB[NTILE * KROW]; // 10 KB
  __shared__ __attribute__((aligned(16))) int    ktab[KPADMAX];       // 3.2 KB
  __shared__ __attribute__((aligned(16))) float4 pp[COUT_];           // 1 KB

  const int t    = threadIdx.x;
  const int lane = t & 31;
  const int wave = t >> 5;

  const int ks  = 3 + 2 * sub_path[1];
  const int ksq = ks * ks;
  const int K   = CIN_ * ksq;
  const int kpad = (K + 31) & ~31;
  const int pad  = ks >> 1;

  const int n_base = blockIdx.x * NTILE;
  const int bb  = n_base >> 16;            // OH*OW = 65536
  const int oh  = (n_base >> 8) & 255;
  const int ow0 = n_base & 255;
  const int ih0 = oh * 2 - pad;

  if (t < COUT_) {                         // fused epilogue constants (float4)
    float a = gamma[t] * rsqrtf(var[t] + 1e-5f);
    pp[t] = make_float4(a, beta[t] - mean[t] * a + b1[t], slope[t], b2[t]);
  }
  // LUT: fold row math (ih clamp + row-valid, block-uniform) into each entry.
  for (int k = t; k < kpad; k += 256) {
    int info = 0;
    if (k < K) {
      int ci  = k / ksq;
      int rem = k - ci * ksq;
      int kh = rem / ks;
      int kw = rem - kh * ks;
      int ih = ih0 + kh;
      int ok = ((unsigned)ih < HW_) ? 1 : 0;
      int ihc = ih < 0 ? 0 : (ih > HW_ - 1 ? HW_ - 1 : ih);
      info = ((ci * HW_ + ihc) * HW_) | (kw << 23) | (ok << 26);
    }
    ktab[k] = info;
  }

  StageCtx sc;
  sc.xb    = x + (size_t)bb * CIN_ * HW_ * HW_;
  sc.ktab  = ktab;
  sc.ns    = t >> 1;
  sc.kbase = (t & 1) * 16;
  sc.iwb   = (ow0 + sc.ns) * 2 - pad;

  __syncthreads();
  {                                        // prologue: stage chunk 0
    float v0[16]; uint ok0;
    stage_load(sc, 0, v0, ok0);
    stage_store(tileA, sc, v0, ok0);
  }

  v8f acc0 = {}, acc1 = {}, acc2 = {}, acc3 = {};
  const int mA   = lane & 15;
  const int khiA = (lane & 16) ? 8 : 0;    // A-frag K sub-offset (ISA layout)
  const int nloc = wave * 16 + (lane & 15);
  const int bsel = (lane & 16) ? 8 : 0;    // B-frag K sub-offset

  for (int kc = 0; kc < kpad; kc += 32) {
    ushort* cur = (kc & 32) ? tileB : tileA;
    ushort* nxt = (kc & 32) ? tileA : tileB;
    const bool more = (kc + 32) < kpad;
    __syncthreads();                       // cur staged; prior reads retired

    // 1) frag loads FIRST (B from LDS, four distinct A frags from global)
    BFrag b, a0, a1, a2, a3;
    const ushort* bp = cur + nloc * KROW + bsel;
    b.h[0] = *(const ushort8*)(bp);
    b.h[1] = *(const ushort8*)(bp + 16);
    const ushort* ap = wbf + mA * KPADMAX + kc + khiA;
    a0.h[0] = *(const ushort8*)(ap);
    a0.h[1] = *(const ushort8*)(ap + 16);
    a1.h[0] = *(const ushort8*)(ap + 16 * KPADMAX);
    a1.h[1] = *(const ushort8*)(ap + 16 * KPADMAX + 16);
    a2.h[0] = *(const ushort8*)(ap + 32 * KPADMAX);
    a2.h[1] = *(const ushort8*)(ap + 32 * KPADMAX + 16);
    a3.h[0] = *(const ushort8*)(ap + 48 * KPADMAX);
    a3.h[1] = *(const ushort8*)(ap + 48 * KPADMAX + 16);

    // 2) issue next-chunk staging loads (values parked in regs)
    float vals[16]; uint okbits = 0;
    if (more) stage_load(sc, kc + 32, vals, okbits);

    // 3) WMMAs only depend on the early loads -> partial loadcnt wait
    acc0 = __builtin_amdgcn_wmma_f32_16x16x32_bf16(false, a0.v, false, b.v,
                                                   (short)0, acc0, false, false);
    acc1 = __builtin_amdgcn_wmma_f32_16x16x32_bf16(false, a1.v, false, b.v,
                                                   (short)0, acc1, false, false);
    acc2 = __builtin_amdgcn_wmma_f32_16x16x32_bf16(false, a2.v, false, b.v,
                                                   (short)0, acc2, false, false);
    acc3 = __builtin_amdgcn_wmma_f32_16x16x32_bf16(false, a3.v, false, b.v,
                                                   (short)0, acc3, false, false);

    // 4) pack + LDS store of the next chunk (the only full-load wait)
    if (more) stage_store(nxt, sc, vals, okbits);
  }

  // epilogue: one ds_load_b128 of constants per channel, fully branchless
  const int hi = (lane & 16) ? 8 : 0;
  const int ow = ow0 + wave * 16 + (lane & 15);
  float* ob = out + ((size_t)bb * COUT_ << 16) + (oh << 8) + ow;
#pragma unroll
  for (int mf = 0; mf < 4; ++mf) {
    v8f acc = mf == 0 ? acc0 : (mf == 1 ? acc1 : (mf == 2 ? acc2 : acc3));
#pragma unroll
    for (int r = 0; r < 8; ++r) {
      int c = mf * 16 + hi + r;
      float4 p = pp[c];
      float v = acc[r] * p.x + p.y;
      v = (v >= 0.f) ? v : v * p.z;        // slope in reg -> v_cndmask
      v += p.w;
      ob[(size_t)c << 16] = v;
    }
  }
}

// ---------------- 4) loss pass-through ------------------------------------
__global__ void stem_loss_kernel(const float* __restrict__ loss,
                                 float* __restrict__ out_loss) {
  out_loss[0] = loss[0];
}

extern "C" void kernel_launch(void* const* d_in, const int* in_sizes, int n_in,
                              void* d_out, int out_size, void* d_ws, size_t ws_size,
                              hipStream_t stream) {
  const float* x        = (const float*)d_in[0];
  const float* loss     = (const float*)d_in[1];
  const int*   sub_path = (const int*)d_in[2];
  const float* weight   = (const float*)d_in[3];
  const float* gamma    = (const float*)d_in[4];
  const float* beta     = (const float*)d_in[5];
  const float* mean     = (const float*)d_in[6];
  const float* var      = (const float*)d_in[7];
  const float* b1       = (const float*)d_in[8];
  const float* slope    = (const float*)d_in[9];
  const float* b2       = (const float*)d_in[10];
  float* out = (float*)d_out;

  float*  scale_p = (float*)d_ws;
  ushort* wbf     = (ushort*)((char*)d_ws + 16);   // 64*800 bf16 = 100 KB

  stem_scale_kernel<<<1, 256, 0, stream>>>(weight, sub_path, scale_p);
  stem_pack_kernel<<<(COUT_ * KPADMAX + 255) / 256, 256, 0, stream>>>(
      weight, sub_path, scale_p, wbf);

  stem_conv_kernel<<<8192, 256, 0, stream>>>(x, wbf, sub_path, gamma, beta,
                                             mean, var, b1, slope, b2, out);

  stem_loss_kernel<<<1, 1, 0, stream>>>(
      loss, out + (size_t)B_ * COUT_ * OHW_ * OHW_);
}
